// trajectory2seq_83056077570408
// MI455X (gfx1250) — matrix operand order repaired and checked by reading, not compiled, not run on hardware
//
#include <hip/hip_runtime.h>
#include <hip/hip_bf16.h>
#include <math.h>
#include <stdint.h>

// ---------------- problem constants ----------------
#define BDIM 128
#define SIN  1024
#define TOUT 128
#define HD   512
#define VOC  32
#define GHD  (3 * HD)     // 1536 gate rows
#define NWG  16           // persistent workgroups
#define TPB  256          // 8 wave32 per WG
#define NB   32           // H-columns owned per WG (16*32 = 512)

typedef __attribute__((ext_vector_type(16))) __bf16 v16bf;
typedef __attribute__((ext_vector_type(8)))  __bf16 v8bf;
typedef __attribute__((ext_vector_type(8)))  float  v8f;

// ---------------- async-to-LDS staging (guarded) ----------------
#if defined(__HIP_DEVICE_COMPILE__) && defined(__has_builtin)
#  if __has_builtin(__builtin_amdgcn_global_load_async_to_lds_b128)
#    define USE_ASYNC_LDS 1
#  endif
#endif
#ifndef USE_ASYNC_LDS
#  define USE_ASYNC_LDS 0
#endif

// Builtin prototype (from hipcc diagnostic): param0 = int4 in AS1 (global),
// param1 = int4 in AS3 (LDS), then imm offset, imm cpol.
typedef int gv4i __attribute__((vector_size(16)));
#define TO_AS1_V4(p) ((__attribute__((address_space(1))) gv4i*)(uintptr_t)(p))
#define TO_AS3_V4(p) ((__attribute__((address_space(3))) gv4i*)(uintptr_t)(p))

__device__ __forceinline__ void stage_chunk16(__bf16* dst_lds, const __bf16* src_glb) {
#if USE_ASYNC_LDS
  __builtin_amdgcn_global_load_async_to_lds_b128(TO_AS1_V4(src_glb), TO_AS3_V4(dst_lds), 0, 0);
#else
  *reinterpret_cast<v8bf*>(dst_lds) = *reinterpret_cast<const v8bf*>(src_glb);
#endif
}
__device__ __forceinline__ void stage_fence() {
#if USE_ASYNC_LDS
  asm volatile("s_wait_asynccnt 0" ::: "memory");
#endif
}

// Stage 16 contiguous rows (16*HD bf16 = 16KB) into this wave's LDS region.
__device__ __forceinline__ void stage_rows(__bf16* dst, const __bf16* src, int lane) {
  for (int i = lane; i < (16 * HD) / 8; i += 32)
    stage_chunk16(dst + i * 8, src + i * 8);
  stage_fence();
}
// Stage 16 gathered rows: row r comes from tbl[rows[mbase+r]].
__device__ __forceinline__ void stage_rows_gather(__bf16* dst, const __bf16* tbl,
                                                  const int* rows, int mbase, int lane) {
  for (int i = lane; i < (16 * HD) / 8; i += 32) {
    const int r = i >> 6;            // HD/8 = 64 chunks per row
    const int c = (i & 63) * 8;
    stage_chunk16(dst + i * 8, tbl + (size_t)rows[mbase + r] * HD + c);
  }
  stage_fence();
}

// ---------------- WMMA fragment loads (layouts per ISA 7.12.2) ----------------
// A: 16x32 bf16, rows staged in LDS (ld = HD). m = lane&15;
// lanes<16 hold K {0..7,16..23}, lanes>=16 hold K {8..15,24..31}.
__device__ __forceinline__ v16bf load_a_frag(const __bf16* A16, int kk, int lane) {
  const int m  = lane & 15;
  const int kb = (lane >> 4) << 3;                 // 0 or 8
  const __bf16* p = A16 + (size_t)m * HD + kk + kb;
  union { v16bf v; v8bf h[2]; } u;
  u.h[0] = *reinterpret_cast<const v8bf*>(p);
  u.h[1] = *reinterpret_cast<const v8bf*>(p + 16);
  return u.v;
}
// B: 32x16 bf16 from weight W[j][k] (row-major, ld=HD), n = lane&15,
// lanes<16 hold K {0..7,16..23}, lanes>=16 hold K {8..15,24..31} (two 16B loads).
__device__ __forceinline__ v16bf load_b_frag(const __bf16* W, int ntile, int kk, int lane) {
  const int n  = ntile * 16 + (lane & 15);
  const int ko = kk + ((lane >> 4) << 3);          // 0 or 8
  const __bf16* p = W + (size_t)n * HD + ko;
  union { v16bf v; v8bf h[2]; } u;
  u.h[0] = *reinterpret_cast<const v8bf*>(p);
  u.h[1] = *reinterpret_cast<const v8bf*>(p + 16);
  return u.v;
}

// C += A(16x512) * W^T slice(512x32), K accumulated in 16 steps of 32.
__device__ __forceinline__ void gemm_acc(const __bf16* A16, const __bf16* Wg,
                                         v8f& acc0, v8f& acc1, int lane) {
#pragma unroll 4
  for (int kk = 0; kk < HD; kk += 32) {
    const v16bf a  = load_a_frag(A16, kk, lane);
    const v16bf b0 = load_b_frag(Wg, 0, kk, lane);
    acc0 = __builtin_amdgcn_wmma_f32_16x16x32_bf16(false, a, false, b0, (short)0, acc0, false, false);
    const v16bf b1 = load_b_frag(Wg, 1, kk, lane);
    acc1 = __builtin_amdgcn_wmma_f32_16x16x32_bf16(false, a, false, b1, (short)0, acc1, false, false);
  }
}

// ---------------- misc device helpers ----------------
__device__ __forceinline__ float sigmoidf_(float x) { return 1.f / (1.f + __expf(-x)); }

__device__ __forceinline__ void grid_barrier(unsigned* bar, int slot) {
  __threadfence();
  __syncthreads();
  if (threadIdx.x == 0) {
    __hip_atomic_fetch_add(&bar[slot], 1u, __ATOMIC_ACQ_REL, __HIP_MEMORY_SCOPE_AGENT);
    while (__hip_atomic_load(&bar[slot], __ATOMIC_ACQUIRE, __HIP_MEMORY_SCOPE_AGENT) < (unsigned)NWG)
      __builtin_amdgcn_s_sleep(1);
  }
  __syncthreads();
}

// GRU fusion for the standard (H-input) layers. acc arrays: ra/za = x+h gate sums,
// xn = x-part of n gate, hn = h-part of n gate. C layout: n = lane&15, m = v + 8*(lane>=16).
__device__ __forceinline__ void gru_fuse_std(
    v8f (&ra)[2], v8f (&za)[2], v8f (&xn)[2], v8f (&hn)[2],
    const float* __restrict__ bih, const float* __restrict__ bhh,
    const float* __restrict__ Hf_old, float* __restrict__ Hf_new,
    __bf16* __restrict__ Hb_new,
    __bf16* enc_t, size_t enc_bstride,
    int c0, int mbase, int lane) {
#pragma unroll
  for (int nt = 0; nt < 2; ++nt) {
    const int j = c0 + nt * 16 + (lane & 15);
    const float brz = bih[j] + bhh[j];
    const float bzz = bih[HD + j] + bhh[HD + j];
    const float bnx = bih[2 * HD + j];
    const float bnh = bhh[2 * HD + j];
#pragma unroll
    for (int v = 0; v < 8; ++v) {
      const int b = mbase + v + ((lane >> 4) << 3);
      const float r = sigmoidf_(ra[nt][v] + brz);
      const float z = sigmoidf_(za[nt][v] + bzz);
      const float n = tanhf(xn[nt][v] + bnx + r * (hn[nt][v] + bnh));
      const float hold = Hf_old[(size_t)b * HD + j];
      const float hnew = (1.f - z) * n + z * hold;
      Hf_new[(size_t)b * HD + j] = hnew;
      const __bf16 hb = (__bf16)hnew;
      Hb_new[(size_t)b * HD + j] = hb;
      if (enc_t) enc_t[(size_t)b * enc_bstride + j] = hb;
    }
  }
}

// ---------------- kernels ----------------
__global__ void k_init(unsigned* bar, unsigned* hz, unsigned hwords, int* tok) {
  const unsigned gtid   = blockIdx.x * blockDim.x + threadIdx.x;
  const unsigned stride = gridDim.x * blockDim.x;
  for (unsigned i = gtid; i < 4096u; i += stride) bar[i] = 0u;
  for (unsigned i = gtid; i < hwords; i += stride) hz[i] = 0u;
  if (gtid < BDIM) tok[gtid] = 0;
}

__global__ void k_f32_to_bf16(const float* __restrict__ src, __bf16* __restrict__ dst, int n) {
  const int i = blockIdx.x * blockDim.x + threadIdx.x;
  if (i < n) dst[i] = (__bf16)src[i];
}

// Persistent 2-layer GRU encoder. 16 WGs, each owns a 32-wide column slice of H.
__global__ __launch_bounds__(TPB) void k_encoder(
    const float* __restrict__ x,        // (B, S, 2)
    const float* __restrict__ Wih0,     // (3H, 2) fp32 (tiny, used scalar)
    const float* __restrict__ bih0, const float* __restrict__ bhh0,
    const float* __restrict__ bih1, const float* __restrict__ bhh1,
    const __bf16* __restrict__ Whh0b, const __bf16* __restrict__ Wih1b,
    const __bf16* __restrict__ Whh1b,
    __bf16* __restrict__ H0b, float* __restrict__ H0f,
    __bf16* __restrict__ H1b, float* __restrict__ H1f,
    __bf16* __restrict__ enc,           // (B, S, H) bf16
    unsigned* __restrict__ bar) {
  extern __shared__ __bf16 smem[];
  const int lane  = threadIdx.x & 31;
  const int wave  = threadIdx.x >> 5;
  const int mbase = wave * 16;
  const int c0    = blockIdx.x * NB;
  __bf16* myA = smem + (size_t)wave * 16 * HD;
  int slot = 0;

#pragma unroll 1
  for (int t = 0; t < SIN; ++t) {
    const int rd = t & 1, wr = rd ^ 1;

    // ---------- layer 0: gi from x (K=2, scalar), gh via WMMA over Whh0 ----------
    {
      stage_rows(myA, H0b + ((size_t)rd * BDIM + mbase) * HD, lane);
      v8f rh[2] = {}, zh[2] = {}, nh[2] = {};
      gemm_acc(myA, Whh0b + (size_t)(0 * HD + c0) * HD, rh[0], rh[1], lane);
      gemm_acc(myA, Whh0b + (size_t)(1 * HD + c0) * HD, zh[0], zh[1], lane);
      gemm_acc(myA, Whh0b + (size_t)(2 * HD + c0) * HD, nh[0], nh[1], lane);
      const float* Hf_old = H0f + (size_t)rd * BDIM * HD;
      float*       Hf_new = H0f + (size_t)wr * BDIM * HD;
      __bf16*      Hb_new = H0b + (size_t)wr * BDIM * HD;
#pragma unroll
      for (int nt = 0; nt < 2; ++nt) {
        const int j = c0 + nt * 16 + (lane & 15);
        const float br = bih0[j],          bhr = bhh0[j];
        const float bz = bih0[HD + j],     bhz = bhh0[HD + j];
        const float bn = bih0[2 * HD + j], bhn = bhh0[2 * HD + j];
        const float wr0 = Wih0[((size_t)(0 * HD + j)) * 2 + 0];
        const float wr1 = Wih0[((size_t)(0 * HD + j)) * 2 + 1];
        const float wz0 = Wih0[((size_t)(1 * HD + j)) * 2 + 0];
        const float wz1 = Wih0[((size_t)(1 * HD + j)) * 2 + 1];
        const float wn0 = Wih0[((size_t)(2 * HD + j)) * 2 + 0];
        const float wn1 = Wih0[((size_t)(2 * HD + j)) * 2 + 1];
#pragma unroll
        for (int v = 0; v < 8; ++v) {
          const int b = mbase + v + ((lane >> 4) << 3);
          const float x0 = x[((size_t)b * SIN + t) * 2 + 0];
          const float x1 = x[((size_t)b * SIN + t) * 2 + 1];
          const float r = sigmoidf_(x0 * wr0 + x1 * wr1 + br + rh[nt][v] + bhr);
          const float z = sigmoidf_(x0 * wz0 + x1 * wz1 + bz + zh[nt][v] + bhz);
          const float n = tanhf(x0 * wn0 + x1 * wn1 + bn + r * (nh[nt][v] + bhn));
          const float hold = Hf_old[(size_t)b * HD + j];
          const float hnew = (1.f - z) * n + z * hold;
          Hf_new[(size_t)b * HD + j] = hnew;
          Hb_new[(size_t)b * HD + j] = (__bf16)hnew;
        }
      }
    }
    grid_barrier(bar, slot++);

    // ---------- layer 1: X = layer0 new h ----------
    {
      v8f ra[2] = {}, za[2] = {}, xn[2] = {}, hn[2] = {};
      stage_rows(myA, H0b + ((size_t)wr * BDIM + mbase) * HD, lane);   // X
      gemm_acc(myA, Wih1b + (size_t)(0 * HD + c0) * HD, ra[0], ra[1], lane);
      gemm_acc(myA, Wih1b + (size_t)(1 * HD + c0) * HD, za[0], za[1], lane);
      gemm_acc(myA, Wih1b + (size_t)(2 * HD + c0) * HD, xn[0], xn[1], lane);
      stage_rows(myA, H1b + ((size_t)rd * BDIM + mbase) * HD, lane);   // H
      gemm_acc(myA, Whh1b + (size_t)(0 * HD + c0) * HD, ra[0], ra[1], lane);
      gemm_acc(myA, Whh1b + (size_t)(1 * HD + c0) * HD, za[0], za[1], lane);
      gemm_acc(myA, Whh1b + (size_t)(2 * HD + c0) * HD, hn[0], hn[1], lane);
      gru_fuse_std(ra, za, xn, hn, bih1, bhh1,
                   H1f + (size_t)rd * BDIM * HD, H1f + (size_t)wr * BDIM * HD,
                   H1b + (size_t)wr * BDIM * HD,
                   enc + (size_t)t * HD, (size_t)SIN * HD,
                   c0, mbase, lane);
    }
    grid_barrier(bar, slot++);
  }
}

// Persistent greedy decoder with attention.
__global__ __launch_bounds__(TPB) void k_decoder(
    const float* __restrict__ bih0, const float* __restrict__ bhh0,
    const float* __restrict__ bih1, const float* __restrict__ bhh1,
    const float* __restrict__ bq,   const float* __restrict__ bc,
    const __bf16* __restrict__ Wih0b, const __bf16* __restrict__ Whh0b,
    const __bf16* __restrict__ Wih1b, const __bf16* __restrict__ Whh1b,
    const __bf16* __restrict__ Wqb,   const __bf16* __restrict__ Wcb,   // Wc: (V, 2H)
    const __bf16* __restrict__ embb,  // (V, H)
    __bf16* __restrict__ H0b, float* __restrict__ H0f,
    __bf16* __restrict__ H1b, float* __restrict__ H1f,
    const __bf16* __restrict__ enc,   // (B, S, H)
    __bf16* __restrict__ qb,          // (B, H)
    int* __restrict__ tok,
    float* __restrict__ out_vec,      // (B, T, V)
    float* __restrict__ out_hidden,   // (2, B, H)
    float* __restrict__ out_attn,     // (B, S, T)
    unsigned* __restrict__ bar) {
  extern __shared__ __bf16 smem[];
  __shared__ float qs[HD];
  __shared__ float wsh[SIN];
  __shared__ float ctxs[HD];
  __shared__ float red[TPB];
  __shared__ float lgs[VOC];

  const int lane  = threadIdx.x & 31;
  const int wave  = threadIdx.x >> 5;
  const int mbase = wave * 16;
  const int c0    = blockIdx.x * NB;
  __bf16* myA = smem + (size_t)wave * 16 * HD;
  int slot = 0;

#pragma unroll 1
  for (int d = 0; d < TOUT; ++d) {
    const int rd = d & 1, wr = rd ^ 1;

    // ---- GRU layer 0: X = embedding[tok] (gathered) ----
    {
      v8f ra[2] = {}, za[2] = {}, xn[2] = {}, hn[2] = {};
      stage_rows_gather(myA, embb, tok, mbase, lane);
      gemm_acc(myA, Wih0b + (size_t)(0 * HD + c0) * HD, ra[0], ra[1], lane);
      gemm_acc(myA, Wih0b + (size_t)(1 * HD + c0) * HD, za[0], za[1], lane);
      gemm_acc(myA, Wih0b + (size_t)(2 * HD + c0) * HD, xn[0], xn[1], lane);
      stage_rows(myA, H0b + ((size_t)rd * BDIM + mbase) * HD, lane);
      gemm_acc(myA, Whh0b + (size_t)(0 * HD + c0) * HD, ra[0], ra[1], lane);
      gemm_acc(myA, Whh0b + (size_t)(1 * HD + c0) * HD, za[0], za[1], lane);
      gemm_acc(myA, Whh0b + (size_t)(2 * HD + c0) * HD, hn[0], hn[1], lane);
      gru_fuse_std(ra, za, xn, hn, bih0, bhh0,
                   H0f + (size_t)rd * BDIM * HD, H0f + (size_t)wr * BDIM * HD,
                   H0b + (size_t)wr * BDIM * HD, (__bf16*)nullptr, 0,
                   c0, mbase, lane);
    }
    grid_barrier(bar, slot++);

    // ---- GRU layer 1: X = h0n ----
    {
      v8f ra[2] = {}, za[2] = {}, xn[2] = {}, hn[2] = {};
      stage_rows(myA, H0b + ((size_t)wr * BDIM + mbase) * HD, lane);
      gemm_acc(myA, Wih1b + (size_t)(0 * HD + c0) * HD, ra[0], ra[1], lane);
      gemm_acc(myA, Wih1b + (size_t)(1 * HD + c0) * HD, za[0], za[1], lane);
      gemm_acc(myA, Wih1b + (size_t)(2 * HD + c0) * HD, xn[0], xn[1], lane);
      stage_rows(myA, H1b + ((size_t)rd * BDIM + mbase) * HD, lane);
      gemm_acc(myA, Whh1b + (size_t)(0 * HD + c0) * HD, ra[0], ra[1], lane);
      gemm_acc(myA, Whh1b + (size_t)(1 * HD + c0) * HD, za[0], za[1], lane);
      gemm_acc(myA, Whh1b + (size_t)(2 * HD + c0) * HD, hn[0], hn[1], lane);
      gru_fuse_std(ra, za, xn, hn, bih1, bhh1,
                   H1f + (size_t)rd * BDIM * HD, H1f + (size_t)wr * BDIM * HD,
                   H1b + (size_t)wr * BDIM * HD, (__bf16*)nullptr, 0,
                   c0, mbase, lane);
    }
    grid_barrier(bar, slot++);

    // ---- q = h1n @ Wq^T + bq (WMMA), write bf16 q ----
    {
      v8f qa[2] = {};
      stage_rows(myA, H1b + ((size_t)wr * BDIM + mbase) * HD, lane);
      gemm_acc(myA, Wqb + (size_t)c0 * HD, qa[0], qa[1], lane);
#pragma unroll
      for (int nt = 0; nt < 2; ++nt) {
        const int j = c0 + nt * 16 + (lane & 15);
        const float bqj = bq[j];
#pragma unroll
        for (int v = 0; v < 8; ++v) {
          const int b = mbase + v + ((lane >> 4) << 3);
          qb[(size_t)b * HD + j] = (__bf16)(qa[nt][v] + bqj);
        }
      }
    }
    grid_barrier(bar, slot++);

    // ---- attention + logits + argmax. WG owns batches b0..b0+7 (memory-bound -> VALU) ----
    {
      const int b0 = blockIdx.x * (BDIM / NWG);
      for (int bi = 0; bi < BDIM / NWG; ++bi) {
        const int b = b0 + bi;
        for (int h = threadIdx.x; h < HD; h += TPB) qs[h] = (float)qb[(size_t)b * HD + h];
        __syncthreads();
        // att[b, s] = q[b] . enc[b, s, :]
        float av[SIN / TPB];
#pragma unroll
        for (int si = 0; si < SIN / TPB; ++si) {
          const int s = threadIdx.x + si * TPB;
          const __bf16* e = enc + ((size_t)b * SIN + s) * HD;
          float acc = 0.f;
#pragma unroll 2
          for (int k = 0; k < HD; k += 8) {
            const v8bf ev = *reinterpret_cast<const v8bf*>(e + k);
#pragma unroll
            for (int u = 0; u < 8; ++u) acc += (float)ev[u] * qs[k + u];
          }
          av[si] = acc;
        }
        // softmax over s (row fully owned by this WG)
        float m = fmaxf(fmaxf(av[0], av[1]), fmaxf(av[2], av[3]));
        red[threadIdx.x] = m;
        __syncthreads();
        for (int o = TPB / 2; o > 0; o >>= 1) {
          if (threadIdx.x < o) red[threadIdx.x] = fmaxf(red[threadIdx.x], red[threadIdx.x + o]);
          __syncthreads();
        }
        const float rowmax = red[0];
        __syncthreads();
        float esum = 0.f;
#pragma unroll
        for (int si = 0; si < SIN / TPB; ++si) { av[si] = __expf(av[si] - rowmax); esum += av[si]; }
        red[threadIdx.x] = esum;
        __syncthreads();
        for (int o = TPB / 2; o > 0; o >>= 1) {
          if (threadIdx.x < o) red[threadIdx.x] += red[threadIdx.x + o];
          __syncthreads();
        }
        const float inv = 1.f / red[0];
        __syncthreads();
#pragma unroll
        for (int si = 0; si < SIN / TPB; ++si) {
          const int s = threadIdx.x + si * TPB;
          const float w = av[si] * inv;
          wsh[s] = w;
          out_attn[((size_t)b * SIN + s) * TOUT + d] = w;
        }
        __syncthreads();
        // ctx[b, h] = sum_s w[s] * enc[b, s, h]; thread owns h = 2*tid, 2*tid+1 (coalesced)
        {
          const int h2 = threadIdx.x * 2;
          float c0a = 0.f, c1a = 0.f;
          for (int s = 0; s < SIN; ++s) {
            const float w = wsh[s];
            const __bf16* e = enc + ((size_t)b * SIN + s) * HD + h2;
            c0a += w * (float)e[0];
            c1a += w * (float)e[1];
          }
          ctxs[h2] = c0a; ctxs[h2 + 1] = c1a;
        }
        __syncthreads();
        // logits[b, v] = [h1n, ctx] . Wc[v] + bc[v]; V=32, 8 K-partials per v
        {
          const int v    = threadIdx.x & (VOC - 1);
          const int part = threadIdx.x >> 5;      // 0..7
          const int hbeg = part * (HD / 8);       // 64 h each
          float s1 = 0.f;
          const float*  h1  = H1f + ((size_t)wr * BDIM + b) * HD;
          const __bf16* wc1 = Wcb + (size_t)v * (2 * HD);
          for (int h = hbeg; h < hbeg + HD / 8; ++h) {
            s1 += h1[h]   * (float)wc1[h];
            s1 += ctxs[h] * (float)wc1[HD + h];
          }
          red[threadIdx.x] = s1;
          __syncthreads();
          if (part == 0) {
            float tot = bc[v];
#pragma unroll
            for (int p = 0; p < 8; ++p) tot += red[v + p * VOC];
            lgs[v] = tot;
            out_vec[((size_t)b * TOUT + d) * VOC + v] = tot;
          }
          __syncthreads();
          if (threadIdx.x == 0) {
            int best = 0; float bv = lgs[0];
            for (int vv = 1; vv < VOC; ++vv)
              if (lgs[vv] > bv) { bv = lgs[vv]; best = vv; }
            tok[b] = best;
          }
          __syncthreads();
        }
      }
    }
    grid_barrier(bar, slot++);
  }

  // final decoder hidden states -> d_out (buffers index 0 hold step-127 results)
  {
    const int gtid = blockIdx.x * TPB + threadIdx.x;
    for (int i = gtid; i < 2 * BDIM * HD; i += NWG * TPB) {
      const int layer = i / (BDIM * HD);
      const int r     = i - layer * (BDIM * HD);
      out_hidden[i] = layer ? H1f[r] : H0f[r];
    }
  }
}

// ---------------- host ----------------
extern "C" void kernel_launch(void* const* d_in, const int* in_sizes, int n_in,
                              void* d_out, int out_size, void* d_ws, size_t ws_size,
                              hipStream_t stream) {
  (void)in_sizes; (void)n_in; (void)out_size; (void)ws_size;
  const float* x     = (const float*)d_in[0];
  const float* emb   = (const float*)d_in[1];
  const float* eWih0 = (const float*)d_in[2];
  const float* eWhh0 = (const float*)d_in[3];
  const float* ebih0 = (const float*)d_in[4];
  const float* ebhh0 = (const float*)d_in[5];
  const float* eWih1 = (const float*)d_in[6];
  const float* eWhh1 = (const float*)d_in[7];
  const float* ebih1 = (const float*)d_in[8];
  const float* ebhh1 = (const float*)d_in[9];
  const float* dWih0 = (const float*)d_in[10];
  const float* dWhh0 = (const float*)d_in[11];
  const float* dbih0 = (const float*)d_in[12];
  const float* dbhh0 = (const float*)d_in[13];
  const float* dWih1 = (const float*)d_in[14];
  const float* dWhh1 = (const float*)d_in[15];
  const float* dbih1 = (const float*)d_in[16];
  const float* dbhh1 = (const float*)d_in[17];
  const float* Wq    = (const float*)d_in[18];
  const float* bq    = (const float*)d_in[19];
  const float* Wc    = (const float*)d_in[20];
  const float* bc    = (const float*)d_in[21];

  char* ws = (char*)d_ws;
  size_t off = 0;
  auto take = [&](size_t bytes) -> void* {
    void* p = ws + off;
    off += (bytes + 255) & ~(size_t)255;
    return p;
  };

  unsigned* bar = (unsigned*)take(4096 * sizeof(unsigned));
  int* tok      = (int*)take(BDIM * sizeof(int));
  const size_t wbytes = (size_t)GHD * HD * sizeof(__bf16);
  __bf16* Whh0b  = (__bf16*)take(wbytes);
  __bf16* Wih1b  = (__bf16*)take(wbytes);
  __bf16* Whh1b  = (__bf16*)take(wbytes);
  __bf16* dWih0b = (__bf16*)take(wbytes);
  __bf16* dWhh0b = (__bf16*)take(wbytes);
  __bf16* dWih1b = (__bf16*)take(wbytes);
  __bf16* dWhh1b = (__bf16*)take(wbytes);
  __bf16* Wqb    = (__bf16*)take((size_t)HD * HD * sizeof(__bf16));
  __bf16* Wcb    = (__bf16*)take((size_t)VOC * 2 * HD * sizeof(__bf16));
  __bf16* embb   = (__bf16*)take((size_t)VOC * HD * sizeof(__bf16));
  // H state zone (kept contiguous; all sizes multiples of 256B)
  __bf16* H0b = (__bf16*)take((size_t)2 * BDIM * HD * sizeof(__bf16));
  __bf16* H1b = (__bf16*)take((size_t)2 * BDIM * HD * sizeof(__bf16));
  float*  H0f = (float*) take((size_t)2 * BDIM * HD * sizeof(float));
  float*  H1f = (float*) take((size_t)2 * BDIM * HD * sizeof(float));
  const size_t hzone_words =
      ((size_t)2 * BDIM * HD * sizeof(__bf16) * 2 + (size_t)2 * BDIM * HD * sizeof(float) * 2) / 4;
  __bf16* qb  = (__bf16*)take((size_t)BDIM * HD * sizeof(__bf16));
  __bf16* enc = (__bf16*)take((size_t)BDIM * SIN * HD * sizeof(__bf16));

  k_init<<<64, 256, 0, stream>>>(bar, (unsigned*)H0b, (unsigned)hzone_words, tok);

  auto cvt = [&](const float* s, __bf16* dptr, int n) {
    k_f32_to_bf16<<<(n + 255) / 256, 256, 0, stream>>>(s, dptr, n);
  };
  cvt(eWhh0, Whh0b, GHD * HD);
  cvt(eWih1, Wih1b, GHD * HD);
  cvt(eWhh1, Whh1b, GHD * HD);
  cvt(dWih0, dWih0b, GHD * HD);
  cvt(dWhh0, dWhh0b, GHD * HD);
  cvt(dWih1, dWih1b, GHD * HD);
  cvt(dWhh1, dWhh1b, GHD * HD);
  cvt(Wq, Wqb, HD * HD);
  cvt(Wc, Wcb, VOC * 2 * HD);
  cvt(emb, embb, VOC * HD);

  const size_t smem = (size_t)8 * 16 * HD * sizeof(__bf16);  // 128 KB per-WG A staging

  k_encoder<<<NWG, TPB, smem, stream>>>(
      x, eWih0, ebih0, ebhh0, ebih1, ebhh1,
      Whh0b, Wih1b, Whh1b,
      H0b, H0f, H1b, H1f, enc, bar);

  float* out_vec    = (float*)d_out;
  float* out_hidden = out_vec + (size_t)BDIM * TOUT * VOC;
  float* out_attn   = out_hidden + (size_t)2 * BDIM * HD;

  k_decoder<<<NWG, TPB, smem, stream>>>(
      dbih0, dbhh0, dbih1, dbhh1, bq, bc,
      dWih0b, dWhh0b, dWih1b, dWhh1b, Wqb, Wcb, embb,
      H0b, H0f, H1b, H1f, enc, qb, tok,
      out_vec, out_hidden, out_attn, bar + 2048);
}
